// GCN2_64630667870322
// MI455X (gfx1250) — matrix operand order repaired and checked by reading, not compiled, not run on hardware
//
#include <hip/hip_runtime.h>

typedef __attribute__((ext_vector_type(16))) _Float16 v16h;
typedef __attribute__((ext_vector_type(8)))  _Float16 v8h;
typedef __attribute__((ext_vector_type(4)))  _Float16 v4h;
typedef __attribute__((ext_vector_type(8)))  float    v8f;
typedef __attribute__((ext_vector_type(4)))  float    v4f;
typedef __attribute__((ext_vector_type(4)))  int      v4i;

#define WAVES 8  // 256-thread block = 8 wave32

__device__ __forceinline__ v16h make_frag(const _Float16* p, int gap) {
  v8h lo = *(const v8h*)p;
  v8h hi = *(const v8h*)(p + gap);
  v16h r;
#pragma unroll
  for (int i = 0; i < 8; ++i) { r[i] = lo[i]; r[i + 8] = hi[i]; }
  return r;
}

__device__ __forceinline__ v4h zero4h() {
  v4h z;
#pragma unroll
  for (int i = 0; i < 4; ++i) z[i] = (_Float16)0.0f;
  return z;
}

// ---------------------------------------------------------------------------
// Edge-message kernel. Concat K-space: [h (Fhp padded) | e (64) | zero pad].
//   EF16:  edge features come from pre-converted f16 buffer; full tiles use
//          global_load_async_to_lds_b128 (contiguous 2KB per tile) overlapped
//          with the scattered h gather, completed with s_wait_asynccnt.
//   WE16:  write f16 copy of edge features while streaming f32 (layer 1).
// sSrc holds src*Fhp, sDst holds dst*strideAgg (pre-multiplied indices).
// ---------------------------------------------------------------------------
template<int KPAD, int NPAD, bool EF16, bool WE16>
__global__ void __launch_bounds__(256) gnn_msg(
    const _Float16* __restrict__ hsrc, int Fh, int Fhp,
    const float* __restrict__ ef32, const _Float16* __restrict__ ef16,
    _Float16* __restrict__ e16out,
    const int* __restrict__ src, const int* __restrict__ dst,
    const float* __restrict__ W, const float* __restrict__ bias,
    int Fout, float* __restrict__ agg, int strideAgg, long nEdges)
{
  __shared__ __align__(16) _Float16 sW[NPAD][KPAD];       // W^T in padded K space
  __shared__ __align__(16) _Float16 sA[WAVES][16][KPAD];  // gathered tiles
  __shared__ float sB[NPAD];
  __shared__ int   sSrc[WAVES][16];
  __shared__ int   sDst[WAVES][16];

  const int tid  = threadIdx.x;
  const int lane = tid & 31;
  const int wave = tid >> 5;

  // Stage weights (transposed, remapped to padded K space) + bias.
  for (int idx = tid; idx < NPAD * KPAD; idx += 256) {
    int n = idx / KPAD, k = idx % KPAD;
    float w = 0.0f;
    if (n < Fout) {
      if (k < Fh)                          w = W[k * Fout + n];
      else if (k >= Fhp && k < Fhp + 64)   w = W[(k - Fhp + Fh) * Fout + n];
    }
    sW[n][k] = (_Float16)w;
  }
  for (int idx = tid; idx < NPAD; idx += 256)
    sB[idx] = (idx < Fout) ? bias[idx] : 0.0f;

  const long tbase = ((long)blockIdx.x * WAVES + wave) * 16;
  const bool fullTile = (tbase + 16 <= nEdges);
  if (lane < 16) {
    long e = tbase + lane;
    int s = 0, d = 0;
    if (e < nEdges) { s = src[e]; d = dst[e]; }
    sSrc[wave][lane] = s * Fhp;       // pre-multiplied element offsets
    sDst[wave][lane] = d * strideAgg;
  }
  __syncthreads();

  constexpr int KQ = KPAD / 4;
  if (EF16 && fullTile) {
    // Async DMA of the contiguous e-section (16 rows x 128B = 2KB) into LDS.
    uint32_t ldsbase = (uint32_t)(size_t)(&sA[wave][0][0]);
    uint64_t gbase   = (uint64_t)(size_t)(ef16 + tbase * 64);
#pragma unroll
    for (int j = 0; j < 4; ++j) {
      int g   = (j * 32 + lane) * 16;   // byte offset within the 2KB chunk
      int m   = g >> 7;                 // row (128B of e-data per row)
      int keb = g & 127;                // byte offset within row's e-section
      uint32_t ldsoff = ldsbase + (uint32_t)(m * (KPAD * 2) + Fhp * 2 + keb);
      uint64_t ga     = gbase + (uint32_t)g;
      asm volatile("global_load_async_to_lds_b128 %0, %1, off"
                   :: "v"(ldsoff), "v"(ga) : "memory");
    }
    // Scattered h gather (+ zero tail padding) overlapped with the DMA.
    for (int idx = lane; idx < 16 * KQ; idx += 32) {
      int m = idx / KQ, k4 = (idx % KQ) * 4;
      if (k4 >= Fhp && k4 < Fhp + 64) continue;   // async covers e-section
      v4h hv = zero4h();
      if (k4 < Fhp) hv = *(const v4h*)(hsrc + sSrc[wave][m] + k4);
      *(v4h*)(&sA[wave][m][k4]) = hv;
    }
    asm volatile("s_wait_asynccnt 0x0" ::: "memory");
  } else {
    // General vectorized gather path (f32 edge stream and/or partial tiles).
    for (int idx = lane; idx < 16 * KQ; idx += 32) {
      int m  = idx / KQ;
      int k4 = (idx % KQ) * 4;
      long e = tbase + m;
      v4h hv = zero4h();
      if (e < nEdges) {
        if (k4 < Fhp) {
          hv = *(const v4h*)(hsrc + sSrc[wave][m] + k4);
        } else {
          int ke = k4 - Fhp;
          if (ke < 64) {
            if (EF16) {
              hv = *(const v4h*)(ef16 + e * 64 + ke);
            } else {
              v4f ev = *(const v4f*)(ef32 + e * 64 + ke);
#pragma unroll
              for (int i = 0; i < 4; ++i) hv[i] = (_Float16)ev[i];
              if (WE16) *(v4h*)(e16out + e * 64 + ke) = hv;
            }
          }
        }
      }
      *(v4h*)(&sA[wave][m][k4]) = hv;
    }
  }
  __syncthreads();

  const int row = lane & 15;   // A: M index; B: N index; C: column
  const int hh  = lane >> 4;

  // Pull all 8 destination offsets for this lane's C rows (two b128 loads).
  v4i d03 = *(const v4i*)(&sDst[wave][hh * 8]);
  v4i d47 = *(const v4i*)(&sDst[wave][hh * 8 + 4]);
  int doff[8] = {d03[0], d03[1], d03[2], d03[3], d47[0], d47[1], d47[2], d47[3]};

#pragma unroll
  for (int nt = 0; nt < NPAD / 16; ++nt) {
    float bv = sB[nt * 16 + row];
    v8f c;
#pragma unroll
    for (int i = 0; i < 8; ++i) c[i] = bv;

#pragma unroll
    for (int kc = 0; kc < KPAD; kc += 32) {
      v16h a = make_frag(&sA[wave][row][kc + hh * 8], 16);
      v16h b = make_frag(&sW[nt * 16 + row][kc + hh * 8], 16);
      c = __builtin_amdgcn_wmma_f32_16x16x32_f16(false, a, false, b,
                                                 (short)0, c, false, false);
    }

    const int col = nt * 16 + row;
    if (fullTile) {            // wave-uniform scalar branch (hot path)
      if (col < Fout) {        // single divergent mask around 8 atomics
#pragma unroll
        for (int r = 0; r < 8; ++r)
          unsafeAtomicAdd(agg + doff[r] + col, fmaxf(c[r], 0.0f));
      }
    } else {                   // cold partial-tile path
#pragma unroll
      for (int r = 0; r < 8; ++r) {
        if (col < Fout && tbase + (r + hh * 8) < nEdges)
          unsafeAtomicAdd(agg + doff[r] + col, fmaxf(c[r], 0.0f));
      }
    }
  }
}

// ---------------------------------------------------------------------------
// Node-apply kernel: out = relu([h | agg] @ W + b). h padded to Fhp (f16),
// agg padded to Faggp (f32, zero-filled). Output stride strideOut, cols >=
// Fout zero-filled so the next layer's padded reads stay valid.
// ---------------------------------------------------------------------------
template<int KPAD, int NPAD, bool OUT16>
__global__ void __launch_bounds__(256) gnn_apply(
    const _Float16* __restrict__ hprev, int Fh, int Fhp,
    const float* __restrict__ aggin, int Fagg, int Faggp,
    const float* __restrict__ W, const float* __restrict__ bias,
    int Fout, int strideOut, void* __restrict__ outp, long nNodes)
{
  __shared__ __align__(16) _Float16 sW[NPAD][KPAD];
  __shared__ __align__(16) _Float16 sA[WAVES][16][KPAD];
  __shared__ float sB[NPAD];

  const int tid  = threadIdx.x;
  const int lane = tid & 31;
  const int wave = tid >> 5;

  for (int idx = tid; idx < NPAD * KPAD; idx += 256) {
    int n = idx / KPAD, k = idx % KPAD;
    float w = 0.0f;
    if (n < Fout) {
      if (k < Fh)         w = W[k * Fout + n];
      else if (k >= Fhp) {
        int ka = k - Fhp;
        if (ka < Fagg)    w = W[(Fh + ka) * Fout + n];
      }
    }
    sW[n][k] = (_Float16)w;
  }
  for (int idx = tid; idx < NPAD; idx += 256)
    sB[idx] = (idx < Fout) ? bias[idx] : 0.0f;
  __syncthreads();

  const long tbase = ((long)blockIdx.x * WAVES + wave) * 16;
  const bool fullTile = (tbase + 16 <= nNodes);

  constexpr int KQ = KPAD / 4;
  for (int idx = lane; idx < 16 * KQ; idx += 32) {
    int m  = idx / KQ;
    int k4 = (idx % KQ) * 4;
    long node = tbase + m;
    v4h hv = zero4h();
    if (node < nNodes) {
      if (k4 < Fhp) {
        hv = *(const v4h*)(hprev + node * Fhp + k4);
      } else {
        int ka = k4 - Fhp;
        if (ka < Faggp) {
          v4f av = *(const v4f*)(aggin + node * Faggp + ka);
#pragma unroll
          for (int i = 0; i < 4; ++i) hv[i] = (_Float16)av[i];
        }
      }
    }
    *(v4h*)(&sA[wave][m][k4]) = hv;
  }
  __syncthreads();

  const int row = lane & 15;
  const int hh  = lane >> 4;

#pragma unroll
  for (int nt = 0; nt < NPAD / 16; ++nt) {
    float bv = sB[nt * 16 + row];
    v8f c;
#pragma unroll
    for (int i = 0; i < 8; ++i) c[i] = bv;

#pragma unroll
    for (int kc = 0; kc < KPAD; kc += 32) {
      v16h a = make_frag(&sA[wave][row][kc + hh * 8], 16);
      v16h b = make_frag(&sW[nt * 16 + row][kc + hh * 8], 16);
      c = __builtin_amdgcn_wmma_f32_16x16x32_f16(false, a, false, b,
                                                 (short)0, c, false, false);
    }

    const int col = nt * 16 + row;
    if (fullTile) {            // hot path: nNodes % 16 == 0
      if (col < strideOut) {
#pragma unroll
        for (int r = 0; r < 8; ++r) {
          long node = tbase + (r + hh * 8);
          float v = (col < Fout) ? fmaxf(c[r], 0.0f) : 0.0f;
          if (OUT16) ((_Float16*)outp)[node * strideOut + col] = (_Float16)v;
          else       ((float*)outp)[node * strideOut + col]    = v;
        }
      }
    } else {
#pragma unroll
      for (int r = 0; r < 8; ++r) {
        long node = tbase + (r + hh * 8);
        float v = (col < Fout) ? fmaxf(c[r], 0.0f) : 0.0f;
        if (col < strideOut && node < nNodes) {
          if (OUT16) ((_Float16*)outp)[node * strideOut + col] = (_Float16)v;
          else       ((float*)outp)[node * strideOut + col]    = v;
        }
      }
    }
  }
}

// ---------------------------------------------------------------------------
__global__ void f32_to_f16_kernel(const float* __restrict__ in,
                                  _Float16* __restrict__ out, long n) {
  long i = (long)blockIdx.x * blockDim.x + threadIdx.x;
  long stride = (long)gridDim.x * blockDim.x;
  for (; i < n; i += stride) out[i] = (_Float16)in[i];
}

__global__ void zero_f32_kernel(float* __restrict__ p, long n) {
  long i = (long)blockIdx.x * blockDim.x + threadIdx.x;
  long stride = (long)gridDim.x * blockDim.x;
  for (; i < n; i += stride) p[i] = 0.0f;
}

// ---------------------------------------------------------------------------
extern "C" void kernel_launch(void* const* d_in, const int* in_sizes, int n_in,
                              void* d_out, int out_size, void* d_ws, size_t ws_size,
                              hipStream_t stream) {
  const float* nfeats = (const float*)d_in[0];
  const float* efeats = (const float*)d_in[1];
  const int*   src    = (const int*)d_in[2];
  const int*   dst    = (const int*)d_in[3];
  const float* Wm1 = (const float*)d_in[4];  const float* bm1 = (const float*)d_in[5];
  const float* Wa1 = (const float*)d_in[6];  const float* ba1 = (const float*)d_in[7];
  const float* Wm2 = (const float*)d_in[8];  const float* bm2 = (const float*)d_in[9];
  const float* Wa2 = (const float*)d_in[10]; const float* ba2 = (const float*)d_in[11];
  const float* Wm3 = (const float*)d_in[12]; const float* bm3 = (const float*)d_in[13];
  const float* Wa3 = (const float*)d_in[14]; const float* ba3 = (const float*)d_in[15];

  const long N = in_sizes[0] / 64;   // 100000
  const long E = in_sizes[2];        // 3200000

  // Padded strides. h strides must keep Fhp*2 16B-aligned for the async-LDS
  // path (hence SH2 = 32, not 28).
  const int SH0 = 64, SH1 = 56, SH2 = 32;       // f16 node-feature strides
  const int SA1 = 52, SA2 = 28, SA3 = 32;       // f32 agg strides

  char* ws = (char*)d_ws;
  size_t off = 0;
  auto alloc = [&](size_t bytes) -> char* {
    char* p = ws + off;
    off = (off + bytes + 255) & ~(size_t)255;
    return p;
  };
  _Float16* h0 = (_Float16*)alloc(N * SH0 * sizeof(_Float16));
  _Float16* h1 = (_Float16*)alloc(N * SH1 * sizeof(_Float16));
  _Float16* h2 = (_Float16*)alloc(N * SH2 * sizeof(_Float16));
  float* aggBase = (float*)alloc(N * (SA1 + SA2 + SA3) * sizeof(float));
  float* agg1 = aggBase;
  float* agg2 = aggBase + N * SA1;
  float* agg3 = aggBase + N * (SA1 + SA2);

  // Optional f16 edge-feature cache (halves layer-2/3 efeats traffic and
  // enables the async-LDS tile DMA).
  size_t e16bytes = (size_t)E * 64 * sizeof(_Float16);
  bool useE16 = (off + e16bytes) <= ws_size;
  _Float16* e16 = useE16 ? (_Float16*)alloc(e16bytes) : nullptr;

  zero_f32_kernel<<<2048, 256, 0, stream>>>(aggBase, N * (SA1 + SA2 + SA3));
  f32_to_f16_kernel<<<2048, 256, 0, stream>>>(nfeats, h0, N * 64);

  const int eBlocks = (int)(((E + 15) / 16 + WAVES - 1) / WAVES);
  const int nBlocks = (int)(((N + 15) / 16 + WAVES - 1) / WAVES);

  // Layer 1 message: K = 64 + 64 = 128, Fout = 50 (NPAD 64)
  if (useE16)
    gnn_msg<128, 64, false, true><<<eBlocks, 256, 0, stream>>>(
        h0, 64, SH0, efeats, nullptr, e16, src, dst, Wm1, bm1, 50, agg1, SA1, E);
  else
    gnn_msg<128, 64, false, false><<<eBlocks, 256, 0, stream>>>(
        h0, 64, SH0, efeats, nullptr, nullptr, src, dst, Wm1, bm1, 50, agg1, SA1, E);
  // Apply 1: K = 64 + 52 = 116 -> 128, Fout = 50, out stride 56 (f16)
  gnn_apply<128, 64, true><<<nBlocks, 256, 0, stream>>>(
      h0, 64, SH0, agg1, 50, SA1, Wa1, ba1, 50, SH1, h1, N);

  // Layer 2 message: K = 56 + 64 = 120 -> 128, Fout = 25 (NPAD 32)
  if (useE16)
    gnn_msg<128, 32, true, false><<<eBlocks, 256, 0, stream>>>(
        h1, 50, SH1, nullptr, e16, nullptr, src, dst, Wm2, bm2, 25, agg2, SA2, E);
  else
    gnn_msg<128, 32, false, false><<<eBlocks, 256, 0, stream>>>(
        h1, 50, SH1, efeats, nullptr, nullptr, src, dst, Wm2, bm2, 25, agg2, SA2, E);
  // Apply 2: K = 56 + 28 = 84 -> 96, Fout = 25, out stride 32 (f16)
  gnn_apply<96, 32, true><<<nBlocks, 256, 0, stream>>>(
      h1, 50, SH1, agg2, 25, SA2, Wa2, ba2, 25, SH2, h2, N);

  // Layer 3 message: K = 32 + 64 = 96, Fout = 32 (NPAD 32)
  if (useE16)
    gnn_msg<96, 32, true, false><<<eBlocks, 256, 0, stream>>>(
        h2, 25, SH2, nullptr, e16, nullptr, src, dst, Wm3, bm3, 32, agg3, SA3, E);
  else
    gnn_msg<96, 32, false, false><<<eBlocks, 256, 0, stream>>>(
        h2, 25, SH2, efeats, nullptr, nullptr, src, dst, Wm3, bm3, 32, agg3, SA3, E);
  // Apply 3: K = 32 + 32 = 64, Fout = 32, f32 straight to d_out
  gnn_apply<64, 32, false><<<nBlocks, 256, 0, stream>>>(
      h2, 25, SH2, agg3, 32, SA3, Wa3, ba3, 32, 32, d_out, N);
}
